// CMHA_50517405335580
// MI455X (gfx1250) — matrix operand order repaired and checked by reading, not compiled, not run on hardware
//
#include <hip/hip_runtime.h>

typedef __bf16 bf16;
typedef __attribute__((ext_vector_type(16))) __bf16 v16bf;
typedef __attribute__((ext_vector_type(8)))  float  v8f;

union FragBF { v16bf v; bf16 e[16]; };
union FragF  { v8f  v;  float e[8]; };

__device__ __forceinline__ float bf2f(bf16 x) { return (float)x; }
__device__ __forceinline__ bf16  f2bf(float x) { return (bf16)x; }

// Fixed problem dimensions (harness-defined).
constexpr int Bc = 2;      // batch
constexpr int Lc = 2048;   // sequence
constexpr int Dc = 1024;   // d_model
constexpr int Hc = 16;     // heads
constexpr int Cc = 128;    // chunk size
constexpr int Nc = 16;     // num chunks
constexpr int Mc = Bc * Lc;             // 4096 rows
constexpr float INV_SCALE = 1.0f / 32.0f;  // 1/sqrt(D)

// -------------------------------------------------------- async LDS copies
// CDNA5 GLOBAL_LOAD_ASYNC_TO_LDS_* : per-lane DMA global->LDS, ASYNCcnt.
__device__ __forceinline__ void async_b128(void* lds, const void* g) {
  asm volatile("global_load_async_to_lds_b128 %0, %1, off"
               :: "v"((unsigned)(uintptr_t)lds), "v"(g) : "memory");
}
__device__ __forceinline__ void async_b32(void* lds, const void* g) {
  asm volatile("global_load_async_to_lds_b32 %0, %1, off"
               :: "v"((unsigned)(uintptr_t)lds), "v"(g) : "memory");
}
__device__ __forceinline__ void wait_async() {
  asm volatile("s_wait_asynccnt 0x0" ::: "memory");
}

// ---------------------------------------------------------------- conversion
__global__ __launch_bounds__(256) void cvt_f32_bf16(const float* __restrict__ in,
                                                    bf16* __restrict__ out, int n) {
  int i = blockIdx.x * 256 + threadIdx.x;
  if (i < n) out[i] = f2bf(in[i]);
}

// in: [K][N] f32 row-major  ->  out: [N][K] bf16 (transposed)
__global__ __launch_bounds__(256) void cvt_transpose_bf16(const float* __restrict__ in,
                                                          bf16* __restrict__ out,
                                                          int K, int N) {
  int idx = blockIdx.x * 256 + threadIdx.x;
  if (idx < K * N) {
    int k = idx / N, n = idx % N;
    out[(size_t)n * K + k] = f2bf(in[idx]);
  }
}

// ---------------------------------------------------------------- WMMA GEMM
// C[M,N] = A[M,K] @ B[K,N] + bias[N], B supplied TRANSPOSED (Bt = [N][K]).
// Block tile 128x64, 8 waves (wave32), each wave a 16x64 strip (4 C tiles),
// K-step 32.  LDS tiles As [m][k], Bs [n][k]: per-lane fragments are
// contiguous bf16 runs -> 2x ds_load_b128 each.  Staging via async-LDS DMA.
// QKV=1: scatter bf16 into Qb/Kb ([b][h][l][64]) and Vt ([b][h][d][L]).
// QKV=0: write f32 to Cout.
template <int N, int K, int QKV>
__global__ __launch_bounds__(256) void gemm_wmma(
    const bf16* __restrict__ A, const bf16* __restrict__ Bt,
    const float* __restrict__ bias, float* __restrict__ Cout,
    bf16* __restrict__ Qb, bf16* __restrict__ Kb, bf16* __restrict__ Vt) {
  __shared__ bf16 As[128 * 32];   // [m][k]
  __shared__ bf16 Bs[64 * 32];    // [n][k]  (transposed tile)
  const int tid = threadIdx.x, lane = tid & 31, w = tid >> 5;
  const int m0 = blockIdx.x * 128, n0 = blockIdx.y * 64;
  const int kb0a = (lane < 16) ? 0 : 8;    // A-fragment K base (16-bit A layout)
  const int kb0b = (lane < 16) ? 0 : 16;   // unused: B fragments also use kb0a
  const int mrow = 16 * w + (lane & 15);   // this lane's A row in the tile

  FragF acc[4];
  for (int t = 0; t < 4; ++t)
    for (int q = 0; q < 8; ++q) acc[t].e[q] = 0.f;

  for (int kk = 0; kk < K; kk += 32) {
    __syncthreads();
#pragma unroll
    for (int rep = 0; rep < 2; ++rep) {          // A: 128x32 bf16 = 512 x b128
      int it = tid + rep * 256;
      int r = it >> 2, q = it & 3;
      async_b128(&As[it * 8], A + (size_t)(m0 + r) * K + kk + q * 8);
    }
    {                                            // Bt: 64x32 bf16 = 256 x b128
      int n = tid >> 2, q = tid & 3;
      async_b128(&Bs[tid * 8], Bt + (size_t)(n0 + n) * K + kk + q * 8);
    }
    wait_async();
    __syncthreads();

    FragBF af;
#pragma unroll
    for (int e = 0; e < 16; ++e) {
      int vg = e >> 1;
      int kl = ((vg < 4) ? kb0a : (kb0a + 16)) + (vg & 3) * 2 + (e & 1);
      af.e[e] = As[mrow * 32 + kl];
    }
#pragma unroll
    for (int nt = 0; nt < 4; ++nt) {
      FragBF bfr;
#pragma unroll
      for (int e = 0; e < 16; ++e)
        bfr.e[e] = Bs[(nt * 16 + (lane & 15)) * 32 + kb0a + e + (e & 8)];
      acc[nt].v = __builtin_amdgcn_wmma_f32_16x16x32_bf16(
          false, af.v, false, bfr.v, (short)0, acc[nt].v, false, false);
    }
  }

#pragma unroll
  for (int nt = 0; nt < 4; ++nt) {
#pragma unroll
    for (int r = 0; r < 8; ++r) {
      int row = m0 + 16 * w + r + ((lane < 16) ? 0 : 8);
      int col = n0 + nt * 16 + (lane & 15);
      float v = acc[nt].e[r] + bias[col];
      if (QKV) {
        int bIdx = row / Lc, l = row % Lc;
        int which = col / Dc, cd = col % Dc;
        int h = cd >> 6, d = cd & 63;
        if (which == 2) {
          // V stored transposed per head: [b][h][d][L]
          Vt[(((size_t)bIdx * Hc + h) * 64 + d) * Lc + l] = f2bf(v);
        } else {
          bf16* dst = (which == 0) ? Qb : Kb;
          dst[(((size_t)bIdx * Hc + h) * Lc + l) * 64 + d] = f2bf(v);
        }
      } else {
        Cout[(size_t)row * N + col] = v;
      }
    }
  }
}

// ---------------------------------------------------------------- pools
// One block per (b,h,chunk); 64 threads = one hd dim each.
__global__ __launch_bounds__(64) void pool_kernel(
    const bf16* __restrict__ Qb, const bf16* __restrict__ Kb,
    float* __restrict__ qpool, float* __restrict__ kpool) {
  int blk = blockIdx.x;
  int n = blk % Nc, bh = blk / Nc;
  int d = threadIdx.x;
  size_t base = ((size_t)bh * Lc + (size_t)n * Cc) * 64 + d;
  float sq = 0.f, sk = 0.f;
  for (int c = 0; c < Cc; ++c) {
    sq += bf2f(Qb[base + (size_t)c * 64]);
    sk += bf2f(Kb[base + (size_t)c * 64]);
  }
  qpool[(bh * Nc + n) * 64 + d] = sq * (1.f / 128.f);
  kpool[(bh * Nc + n) * 64 + d] = sk * (1.f / 128.f);
}

// ---------------------------------------------------------------- chunk2token
// c2t[b,h,j,c] = qpool[b,h,j] . k[b,h,j*C+c]
__global__ __launch_bounds__(128) void c2t_kernel(
    const float* __restrict__ qpool, const bf16* __restrict__ Kb,
    float* __restrict__ c2t) {
  int blk = blockIdx.x;
  int j = blk % Nc, bh = blk / Nc;
  int c = threadIdx.x;
  const float* qp = qpool + (bh * Nc + j) * 64;
  const bf16* kr = Kb + ((size_t)bh * Lc + (size_t)j * Cc + c) * 64;
  float s = 0.f;
  for (int d = 0; d < 64; ++d) s += qp[d] * bf2f(kr[d]);
  c2t[(bh * Nc + j) * Cc + c] = s;
}

// ---------------------------------------------------------------- attention
// One block per (b,h,query chunk i); 8 waves, each owns 16 query rows.
// Phase 1: t2c (q . kpool) and the diagonal block S=Q.K^T via WMMA;
//          P_diag = exp/causal applied in C-layout, then written bf16 to the
//          LDS region previously holding Q/K (dead after phase 1).
// Phase 2: stream key chunks j, P in WMMA A-fragment layout:
//   j<i : exp(t2c[l,j]*c2t[j,c]/32)   (fully unmasked)
//   j==i: P_diag from LDS             (mask -> score 0 -> exp=1)
//   j>i : 1 everywhere                (mask -> score 0 -> exp=1)
// out = sum_j P_j @ V_j / rowsum(P)  -- matches reference softmax exactly.
// V arrives transposed ([b][h][d][L]) so V-fragment reads are contiguous.
__global__ __launch_bounds__(256) void attn_kernel(
    const bf16* __restrict__ Qb, const bf16* __restrict__ Kb,
    const bf16* __restrict__ Vt, const float* __restrict__ kpool,
    const float* __restrict__ c2t, bf16* __restrict__ attnb) {
  __shared__ bf16 QK[2 * 128 * 64];  // phase1: Qs|Ks ; phase2: P_diag[128][128]
  __shared__ bf16 Vts[64 * 128];     // [d][key] (transposed tile)
  __shared__ float t2cs[128 * 16];
  __shared__ float c2ts[128];
  __shared__ float kps[16 * 64];

  bf16* Qs = QK;                 // [l][d]
  bf16* Ks = QK + 128 * 64;      // [key][d]
  bf16* Pd = QK;                 // [l][key] overlay (after phase 1)

  int blk = blockIdx.x;
  int i = blk % Nc;
  int h = (blk / Nc) % Hc;
  int b = blk / (Nc * Hc);
  int bh = b * Hc + h;
  int tid = threadIdx.x, lane = tid & 31, w = tid >> 5;

  const bf16* Qg = Qb + ((size_t)bh * Lc + (size_t)i * Cc) * 64;
  const bf16* Kg = Kb + ((size_t)bh * Lc + (size_t)i * Cc) * 64;
  const float* kpg = kpool + (size_t)bh * Nc * 64;
#pragma unroll
  for (int rep = 0; rep < 4; ++rep) {   // Q,K tiles: 1024 x b128 each
    int it = tid + rep * 256;
    async_b128(&Qs[it * 8], Qg + it * 8);
    async_b128(&Ks[it * 8], Kg + it * 8);
  }
#pragma unroll
  for (int rep = 0; rep < 4; ++rep) {   // kpool: 1024 floats
    int it = tid + rep * 256;
    async_b32(&kps[it], kpg + it);
  }
  wait_async();
  __syncthreads();

  // token2chunk for this query chunk: t2cs[l][n] = q_l . kpool_n
#pragma unroll
  for (int rep = 0; rep < 8; ++rep) {
    int idx = tid + rep * 256;
    int l = idx >> 4, n = idx & 15;
    float s = 0.f;
    for (int d = 0; d < 64; ++d) s += bf2f(Qs[l * 64 + d]) * kps[n * 64 + d];
    t2cs[idx] = s;
  }

  const int kb0a = (lane < 16) ? 0 : 8;
  const int kb0b = (lane < 16) ? 0 : 16;
  const int ql = 16 * w + (lane & 15);  // this lane's query row (A-layout M)

  // ---- phase 1: diagonal block S = Q.K^T via WMMA (16 wmma per wave) ----
  FragBF qa[2];
#pragma unroll
  for (int step = 0; step < 2; ++step)
#pragma unroll
    for (int e = 0; e < 16; ++e) {
      int vg = e >> 1;
      int kl = ((vg < 4) ? kb0a : (kb0a + 16)) + (vg & 3) * 2 + (e & 1);
      qa[step].e[e] = Qs[ql * 64 + step * 32 + kl];
    }
  FragF sdiag[8];
  for (int g = 0; g < 8; ++g)
    for (int q = 0; q < 8; ++q) sdiag[g].e[q] = 0.f;
#pragma unroll
  for (int g = 0; g < 8; ++g) {
#pragma unroll
    for (int step = 0; step < 2; ++step) {
      FragBF kfr;   // B = K^T : element e <-> d = step*32+kb0a+e+(e&8), col = key
#pragma unroll
      for (int e = 0; e < 16; ++e)
        kfr.e[e] = Ks[(g * 16 + (lane & 15)) * 64 + step * 32 + kb0a + e + (e & 8)];
      sdiag[g].v = __builtin_amdgcn_wmma_f32_16x16x32_bf16(
          false, qa[step].v, false, kfr.v, (short)0, sdiag[g].v, false, false);
    }
  }
  __syncthreads();  // everyone done reading Qs/Ks (and t2cs fully written)

  // exp + causal in C-layout, store bf16 P_diag[l][key] over the Q/K region
#pragma unroll
  for (int g = 0; g < 8; ++g) {
#pragma unroll
    for (int r = 0; r < 8; ++r) {
      int m = r + ((lane < 16) ? 0 : 8);
      int l = 16 * w + m;
      int key = g * 16 + (lane & 15);
      float p = (key <= l) ? __expf(sdiag[g].e[r] * INV_SCALE) : 1.f;
      Pd[l * 128 + key] = f2bf(p);
    }
  }
  __syncthreads();

  // ---- phase 2: stream key chunks, accumulate P@V ----
  FragF acc[4];
  for (int t = 0; t < 4; ++t)
    for (int q = 0; q < 8; ++q) acc[t].e[q] = 0.f;
  float zpart = 0.f;

  for (int j = 0; j < Nc; ++j) {
    __syncthreads();
    // stage V^T tile: 64 rows(d) x 128(key) bf16 = 1024 x b128
    const bf16* Vg = Vt + (size_t)bh * 64 * Lc + (size_t)j * Cc;
    if (j + 1 < Nc)
      __builtin_prefetch(Vt + (size_t)bh * 64 * Lc + (size_t)(j + 1) * Cc, 0, 0);
#pragma unroll
    for (int rep = 0; rep < 4; ++rep) {
      int it = tid + rep * 256;
      int d = it >> 4, q = it & 15;
      async_b128(&Vts[it * 8], Vg + (size_t)d * Lc + q * 8);
    }
    if (tid < 128) async_b32(&c2ts[tid], c2t + (bh * Nc + j) * Cc + tid);
    wait_async();
    __syncthreads();

    float tq = t2cs[ql * 16 + j];

    for (int kg = 0; kg < 4; ++kg) {
      FragBF pa;
      if (j == i) {
        // diagonal P from LDS, already in A-fragment-friendly [l][key] rows
#pragma unroll
        for (int e = 0; e < 16; ++e) {
          int vg = e >> 1;
          int kl = ((vg < 4) ? kb0a : (kb0a + 16)) + (vg & 3) * 2 + (e & 1);
          pa.e[e] = Pd[ql * 128 + kg * 32 + kl];
        }
#pragma unroll
        for (int e = 0; e < 16; ++e) zpart += bf2f(pa.e[e]);
      } else {
#pragma unroll
        for (int e = 0; e < 16; ++e) {
          int vg = e >> 1;
          int kl = ((vg < 4) ? kb0a : (kb0a + 16)) + (vg & 3) * 2 + (e & 1);
          int key = kg * 32 + kl;
          float p = (j < i) ? __expf(tq * c2ts[key] * INV_SCALE) : 1.f;
          zpart += p;
          pa.e[e] = f2bf(p);
        }
      }
#pragma unroll
      for (int nt = 0; nt < 4; ++nt) {
        FragBF vfr;
#pragma unroll
        for (int e = 0; e < 16; ++e)
          vfr.e[e] = Vts[(nt * 16 + (lane & 15)) * 128 + kg * 32 + kb0a + e + (e & 8)];
        acc[nt].v = __builtin_amdgcn_wmma_f32_16x16x32_bf16(
            false, pa.v, false, vfr.v, (short)0, acc[nt].v, false, false);
      }
    }
  }

  // full row sums: lanes m and m+16 cover disjoint halves of each key group
  float zc = zpart + __shfl_xor(zpart, 16, 32);

#pragma unroll
  for (int nt = 0; nt < 4; ++nt) {
#pragma unroll
    for (int r = 0; r < 8; ++r) {
      int row = r + ((lane < 16) ? 0 : 8);     // C-layout row within strip
      float Z = __shfl(zc, row, 32);           // lane 'row' holds rowsum(row)
      int lg = i * Cc + 16 * w + row;
      attnb[((size_t)b * Lc + lg) * Dc + h * 64 + nt * 16 + (lane & 15)] =
          f2bf(acc[nt].e[r] / Z);
    }
  }
}

// ---------------------------------------------------------------- launcher
extern "C" void kernel_launch(void* const* d_in, const int* in_sizes, int n_in,
                              void* d_out, int out_size, void* d_ws,
                              size_t ws_size, hipStream_t stream) {
  (void)in_sizes; (void)n_in; (void)out_size; (void)ws_size;

  const float* x     = (const float*)d_in[0];
  const float* w_qkv = (const float*)d_in[1];
  const float* b_qkv = (const float*)d_in[2];
  const float* w_o   = (const float*)d_in[3];
  const float* b_o   = (const float*)d_in[4];

  char* ws = (char*)d_ws;
  size_t off = 0;
  auto take = [&](size_t bytes) -> void* {
    void* p = ws + off;
    off += (bytes + 255) & ~(size_t)255;
    return p;
  };

  bf16* xb   = (bf16*)take((size_t)Mc * Dc * 2);
  bf16* wqbT = (bf16*)take((size_t)Dc * 3 * Dc * 2);   // [3D][D]
  bf16* wobT = (bf16*)take((size_t)Dc * Dc * 2);       // [D][D] transposed
  bf16* Qb   = (bf16*)take((size_t)Bc * Hc * Lc * 64 * 2);
  bf16* Kb   = (bf16*)take((size_t)Bc * Hc * Lc * 64 * 2);
  bf16* Vt   = (bf16*)take((size_t)Bc * Hc * 64 * Lc * 2);
  float* qpool = (float*)take((size_t)Bc * Hc * Nc * 64 * 4);
  float* kpool = (float*)take((size_t)Bc * Hc * Nc * 64 * 4);
  float* c2t   = (float*)take((size_t)Bc * Hc * Nc * Cc * 4);
  bf16* attnb  = (bf16*)take((size_t)Mc * Dc * 2);

  cvt_f32_bf16<<<(Mc * Dc + 255) / 256, 256, 0, stream>>>(x, xb, Mc * Dc);
  cvt_transpose_bf16<<<(Dc * 3 * Dc + 255) / 256, 256, 0, stream>>>(
      w_qkv, wqbT, Dc, 3 * Dc);
  cvt_transpose_bf16<<<(Dc * Dc + 255) / 256, 256, 0, stream>>>(w_o, wobT, Dc, Dc);

  dim3 g1(Mc / 128, (3 * Dc) / 64);
  gemm_wmma<3 * Dc, Dc, 1><<<g1, 256, 0, stream>>>(xb, wqbT, b_qkv, nullptr,
                                                   Qb, Kb, Vt);

  pool_kernel<<<Bc * Hc * Nc, 64, 0, stream>>>(Qb, Kb, qpool, kpool);
  c2t_kernel<<<Bc * Hc * Nc, 128, 0, stream>>>(qpool, Kb, c2t);

  attn_kernel<<<Bc * Hc * Nc, 256, 0, stream>>>(Qb, Kb, Vt, kpool, c2t, attnb);

  dim3 g2(Mc / 128, Dc / 64);
  gemm_wmma<Dc, Dc, 0><<<g2, 256, 0, stream>>>(attnb, wobT, b_o, (float*)d_out,
                                               nullptr, nullptr, nullptr);
}
